// GINBaseline_19610820673868
// MI455X (gfx1250) — compile-verified, matrix operand order checked by reading, not run on hardware
//
#include <hip/hip_runtime.h>
#include <hip/hip_bf16.h>

typedef __attribute__((ext_vector_type(16))) _Float16 v16h;
typedef __attribute__((ext_vector_type(8)))  _Float16 v8h;
typedef __attribute__((ext_vector_type(4)))  _Float16 v4h;
typedef __attribute__((ext_vector_type(8)))  float    v8f;

#define HDIM 128
#define LDS_STRIDE 136   // 128 + 8 halves pad -> 272B row stride, conflict-free

// ---------------------------------------------------------------------------
// Generic zero fill
// ---------------------------------------------------------------------------
__global__ void gin_fill0(float* __restrict__ p, long n) {
  long i = (long)blockIdx.x * blockDim.x + threadIdx.x;
  if (i < n) p[i] = 0.0f;
}

// ---------------------------------------------------------------------------
// Weight prep: Bt[n][k] = (f16) W[k][n]   (128x128)
// ---------------------------------------------------------------------------
__global__ void gin_w_to_f16t(const float* __restrict__ W, _Float16* __restrict__ Bt) {
  int i = blockIdx.x * blockDim.x + threadIdx.x;  // 0..16383
  int n = i >> 7, k = i & 127;
  Bt[i] = (_Float16)W[k * HDIM + n];
}

// ---------------------------------------------------------------------------
// Edge aggregation: one wave per edge, float4 per lane, L2-resident atomics
// ---------------------------------------------------------------------------
__global__ void gin_edge_agg(const float* __restrict__ h,
                             const int* __restrict__ src,
                             const int* __restrict__ dst,
                             float* __restrict__ agg, int nEdges) {
  int gid  = blockIdx.x * blockDim.x + threadIdx.x;
  int edge = gid >> 5;
  int lane = gid & 31;
  if (edge >= nEdges) return;
  int s = src[edge], d = dst[edge];
  float4 v = ((const float4*)(h + (size_t)s * HDIM))[lane];
  float* out = agg + (size_t)d * HDIM + lane * 4;
  atomicAdd(out + 0, v.x);
  atomicAdd(out + 1, v.y);
  atomicAdd(out + 2, v.z);
  atomicAdd(out + 3, v.w);
}

// ---------------------------------------------------------------------------
// Wave-level 16x128 @ 128x128 strip GEMM using v_wmma_f32_16x16x32_f16.
// A: f16 rows in LDS (stride LDS_STRIDE). Bt: f16, row n holds K contiguously.
// acc[t] covers cols [16t,16t+16).
// ---------------------------------------------------------------------------
__device__ __forceinline__ void gin_wave_gemm(const _Float16* __restrict__ A,
                                              int rowBase,
                                              const _Float16* __restrict__ Bt,
                                              v8f acc[8], int lane) {
  const int r  = lane & 15;
  const int hb = (lane >> 4) * 8;                 // K sub-offset: 0 or 8
  const _Float16* arow = A + (rowBase + r) * LDS_STRIDE;
#pragma unroll
  for (int s = 0; s < 4; ++s) {
    const int k0 = 32 * s + hb;
    v8h a0 = *(const v8h*)(arow + k0);
    v8h a1 = *(const v8h*)(arow + k0 + 16);
    v16h af = __builtin_shufflevector(a0, a1, 0,1,2,3,4,5,6,7,8,9,10,11,12,13,14,15);
#pragma unroll
    for (int t = 0; t < 8; ++t) {
      const _Float16* brow = Bt + (size_t)(t * 16 + r) * HDIM;
      v8h b0 = *(const v8h*)(brow + k0);
      v8h b1 = *(const v8h*)(brow + k0 + 16);
      v16h bf = __builtin_shufflevector(b0, b1, 0,1,2,3,4,5,6,7,8,9,10,11,12,13,14,15);
      acc[t] = __builtin_amdgcn_wmma_f32_16x16x32_f16(
          false, af, false, bf, (short)0, acc[t], false, false);
    }
  }
}

// ---------------------------------------------------------------------------
// GIN MLP: hout = relu( relu((hin+agg)@W1 + b1) @ W2 + b2 )
// 128-row tile per block, 8 waves, 2 chained WMMA GEMMs through one LDS tile.
// Safe to run with hout == agg (rows read before written, block-private rows).
// ---------------------------------------------------------------------------
__global__ __launch_bounds__(256) void gin_mlp_wmma(
    const float* __restrict__ hin, const float* __restrict__ agg,
    const _Float16* __restrict__ W1t, const float* __restrict__ b1,
    const _Float16* __restrict__ W2t, const float* __restrict__ b2,
    float* __restrict__ hout, int nRows) {
  __shared__ _Float16 tile[128 * LDS_STRIDE];
  const int row0 = blockIdx.x * 128;
  const int tid  = threadIdx.x;

  // Stage z = hin + agg as f16
  for (int i = tid; i < 128 * 32; i += 256) {
    const int r = i >> 5, c4 = i & 31;
    const int row = row0 + r;
    float4 hv = make_float4(0.f, 0.f, 0.f, 0.f);
    float4 av = make_float4(0.f, 0.f, 0.f, 0.f);
    if (row < nRows) {
      hv = ((const float4*)(hin + (size_t)row * HDIM))[c4];
      av = ((const float4*)(agg + (size_t)row * HDIM))[c4];
    }
    v4h z;
    z[0] = (_Float16)(hv.x + av.x);
    z[1] = (_Float16)(hv.y + av.y);
    z[2] = (_Float16)(hv.z + av.z);
    z[3] = (_Float16)(hv.w + av.w);
    *(v4h*)(&tile[r * LDS_STRIDE + c4 * 4]) = z;
  }
  __syncthreads();

  const int wave = tid >> 5, lane = tid & 31;
  const int r = lane & 15, hi = lane >> 4;

  v8f acc[8] = {};
  gin_wave_gemm(tile, wave * 16, W1t, acc, lane);
  __syncthreads();   // all waves done reading z; reuse tile

  // relu(acc + b1) -> f16 tile
#pragma unroll
  for (int t = 0; t < 8; ++t) {
    const int col = t * 16 + r;
    const float bb = b1[col];
#pragma unroll
    for (int v = 0; v < 8; ++v) {
      const int rr = wave * 16 + hi * 8 + v;
      float x = acc[t][v] + bb;
      tile[rr * LDS_STRIDE + col] = (_Float16)(x > 0.f ? x : 0.f);
    }
  }
  __syncthreads();

  v8f acc2[8] = {};
  gin_wave_gemm(tile, wave * 16, W2t, acc2, lane);

#pragma unroll
  for (int t = 0; t < 8; ++t) {
    const int col = t * 16 + r;
    const float bb = b2[col];
#pragma unroll
    for (int v = 0; v < 8; ++v) {
      const int rr = row0 + wave * 16 + hi * 8 + v;
      if (rr < nRows) {
        float x = acc2[t][v] + bb;
        hout[(size_t)rr * HDIM + col] = x > 0.f ? x : 0.f;
      }
    }
  }
}

// ---------------------------------------------------------------------------
// Attention gate: warp-per-node dot(h, gate_w)+b, segment max via ordered-uint
// ---------------------------------------------------------------------------
__global__ void gin_gate(const float* __restrict__ h, const float* __restrict__ gw,
                         const float* __restrict__ gb, const int* __restrict__ batch,
                         float* __restrict__ gate, unsigned* __restrict__ mU, int n) {
  int gid  = blockIdx.x * blockDim.x + threadIdx.x;
  int node = gid >> 5, lane = gid & 31;
  if (node >= n) return;
  float4 hv = ((const float4*)(h + (size_t)node * HDIM))[lane];
  float4 wv = ((const float4*)gw)[lane];
  float s = hv.x * wv.x + hv.y * wv.y + hv.z * wv.z + hv.w * wv.w;
#pragma unroll
  for (int o = 16; o > 0; o >>= 1) s += __shfl_xor(s, o, 32);
  if (lane == 0) {
    float gv = s + gb[0];
    gate[node] = gv;
    unsigned b = __float_as_uint(gv);
    unsigned ou = (b & 0x80000000u) ? ~b : (b | 0x80000000u);
    atomicMax(&mU[batch[node]], ou);
  }
}

// a = exp(gate - m[batch]); denom[batch] += a   (gate overwritten in place)
__global__ void gin_exp_denom(float* __restrict__ gate, const unsigned* __restrict__ mU,
                              float* __restrict__ denom, const int* __restrict__ batch,
                              int n) {
  int i = blockIdx.x * blockDim.x + threadIdx.x;
  if (i >= n) return;
  int b = batch[i];
  unsigned mu = mU[b];
  float m = (mu & 0x80000000u) ? __uint_as_float(mu ^ 0x80000000u)
                               : __uint_as_float(~mu);
  float a = __expf(gate[i] - m);
  gate[i] = a;
  atomicAdd(&denom[b], a);
}

// g[batch] += h * (a/denom[batch]) : warp per node
__global__ void gin_wsum(const float* __restrict__ h, const float* __restrict__ a,
                         const float* __restrict__ denom, const int* __restrict__ batch,
                         float* __restrict__ g, int n) {
  int gid  = blockIdx.x * blockDim.x + threadIdx.x;
  int node = gid >> 5, lane = gid & 31;
  if (node >= n) return;
  int b = batch[node];
  float w = a[node] / denom[b];
  float4 hv = ((const float4*)(h + (size_t)node * HDIM))[lane];
  float* gp = g + (size_t)b * HDIM + lane * 4;
  atomicAdd(gp + 0, hv.x * w);
  atomicAdd(gp + 1, hv.y * w);
  atomicAdd(gp + 2, hv.z * w);
  atomicAdd(gp + 3, hv.w * w);
}

// ---------------------------------------------------------------------------
// Head: out = relu(g@W1+b1) @ W2 + b2 ; W2 is 128 x OD (f32, small)
// ---------------------------------------------------------------------------
__global__ __launch_bounds__(256) void gin_head_wmma(
    const float* __restrict__ gmat, const _Float16* __restrict__ W1t,
    const float* __restrict__ b1, const float* __restrict__ W2,
    const float* __restrict__ b2, float* __restrict__ out, int OD, int nRows) {
  __shared__ _Float16 tile[128 * LDS_STRIDE];
  const int row0 = blockIdx.x * 128;
  const int tid  = threadIdx.x;

  for (int i = tid; i < 128 * 32; i += 256) {
    const int r = i >> 5, c4 = i & 31;
    const int row = row0 + r;
    float4 hv = make_float4(0.f, 0.f, 0.f, 0.f);
    if (row < nRows) hv = ((const float4*)(gmat + (size_t)row * HDIM))[c4];
    v4h z;
    z[0] = (_Float16)hv.x; z[1] = (_Float16)hv.y;
    z[2] = (_Float16)hv.z; z[3] = (_Float16)hv.w;
    *(v4h*)(&tile[r * LDS_STRIDE + c4 * 4]) = z;
  }
  __syncthreads();

  const int wave = tid >> 5, lane = tid & 31;
  const int r = lane & 15, hi = lane >> 4;
  v8f acc[8] = {};
  gin_wave_gemm(tile, wave * 16, W1t, acc, lane);
  __syncthreads();

#pragma unroll
  for (int t = 0; t < 8; ++t) {
    const int col = t * 16 + r;
    const float bb = b1[col];
#pragma unroll
    for (int v = 0; v < 8; ++v) {
      const int rr = wave * 16 + hi * 8 + v;
      float x = acc[t][v] + bb;
      tile[rr * LDS_STRIDE + col] = (_Float16)(x > 0.f ? x : 0.f);
    }
  }
  __syncthreads();

  // tiny projection 128 -> OD
  for (int i = tid; i < 128 * OD; i += 256) {
    const int rr = i / OD, c = i - rr * OD;
    const int grow = row0 + rr;
    if (grow < nRows) {
      float s = b2[c];
#pragma unroll 8
      for (int k = 0; k < HDIM; ++k)
        s += (float)tile[rr * LDS_STRIDE + k] * W2[k * OD + c];
      out[(size_t)grow * OD + c] = s;
    }
  }
}

// ---------------------------------------------------------------------------
extern "C" void kernel_launch(void* const* d_in, const int* in_sizes, int n_in,
                              void* d_out, int out_size, void* d_ws, size_t ws_size,
                              hipStream_t stream) {
  const int N = in_sizes[2];          // nodes (batch array length)
  const int E = in_sizes[1] / 2;      // edges
  const int G = out_size / 10;        // 9 cls + 1 reg per graph

  const float* x      = (const float*)d_in[0];
  const int*   src    = (const int*)d_in[1];
  const int*   dst    = (const int*)d_in[1] + E;
  const int*   batch  = (const int*)d_in[2];
  const float* l0_w1  = (const float*)d_in[3];
  const float* l0_b1  = (const float*)d_in[4];
  const float* l0_w2  = (const float*)d_in[5];
  const float* l0_b2  = (const float*)d_in[6];
  const float* l1_w1  = (const float*)d_in[7];
  const float* l1_b1  = (const float*)d_in[8];
  const float* l1_w2  = (const float*)d_in[9];
  const float* l1_b2  = (const float*)d_in[10];
  const float* gate_w = (const float*)d_in[11];
  const float* gate_b = (const float*)d_in[12];
  const float* cls_w1 = (const float*)d_in[13];
  const float* cls_b1 = (const float*)d_in[14];
  const float* cls_w2 = (const float*)d_in[15];
  const float* cls_b2 = (const float*)d_in[16];
  const float* reg_w1 = (const float*)d_in[17];
  const float* reg_b1 = (const float*)d_in[18];
  const float* reg_w2 = (const float*)d_in[19];
  const float* reg_b2 = (const float*)d_in[20];
  float* out = (float*)d_out;

  // workspace carve-up
  char* ws = (char*)d_ws;
  size_t off = 0;
  auto carve = [&](size_t bytes) -> void* {
    void* p = ws + off;
    off += (bytes + 255) & ~(size_t)255;
    return p;
  };
  float*     bufA  = (float*)carve((size_t)N * HDIM * 4);  // agg / h2
  float*     bufB  = (float*)carve((size_t)N * HDIM * 4);  // h1
  float*     gateB = (float*)carve((size_t)N * 4);
  unsigned*  mU    = (unsigned*)carve((size_t)G * 4);
  float*     denom = (float*)carve((size_t)G * 4);
  float*     gvec  = (float*)carve((size_t)G * HDIM * 4);
  _Float16*  Wt[6];
  for (int i = 0; i < 6; ++i) Wt[i] = (_Float16*)carve(HDIM * HDIM * 2);

  const int TPB = 256;
  const long nodeFeat = (long)N * HDIM;
  const int  fillBlk  = (int)((nodeFeat + TPB - 1) / TPB);
  const int  waveBlk  = (int)(((long)N * 32 + TPB - 1) / TPB);
  const int  rowBlk   = (N + 127) / 128;

  // Weight prep (f32 -> f16 transposed)
  gin_w_to_f16t<<<64, TPB, 0, stream>>>(l0_w1, Wt[0]);
  gin_w_to_f16t<<<64, TPB, 0, stream>>>(l0_w2, Wt[1]);
  gin_w_to_f16t<<<64, TPB, 0, stream>>>(l1_w1, Wt[2]);
  gin_w_to_f16t<<<64, TPB, 0, stream>>>(l1_w2, Wt[3]);
  gin_w_to_f16t<<<64, TPB, 0, stream>>>(cls_w1, Wt[4]);
  gin_w_to_f16t<<<64, TPB, 0, stream>>>(reg_w1, Wt[5]);

  // Layer 0
  gin_fill0<<<fillBlk, TPB, 0, stream>>>(bufA, nodeFeat);
  gin_edge_agg<<<(int)(((long)E * 32 + TPB - 1) / TPB), TPB, 0, stream>>>(x, src, dst, bufA, E);
  gin_mlp_wmma<<<rowBlk, TPB, 0, stream>>>(x, bufA, Wt[0], l0_b1, Wt[1], l0_b2, bufB, N);

  // Layer 1 (output written in-place over agg buffer)
  gin_fill0<<<fillBlk, TPB, 0, stream>>>(bufA, nodeFeat);
  gin_edge_agg<<<(int)(((long)E * 32 + TPB - 1) / TPB), TPB, 0, stream>>>(bufB, src, dst, bufA, E);
  gin_mlp_wmma<<<rowBlk, TPB, 0, stream>>>(bufB, bufA, Wt[2], l1_b1, Wt[3], l1_b2, bufA, N);

  // Attention readout
  gin_fill0<<<1, TPB, 0, stream>>>((float*)mU, G);
  gin_fill0<<<1, TPB, 0, stream>>>(denom, G);
  gin_fill0<<<(G * HDIM + TPB - 1) / TPB, TPB, 0, stream>>>(gvec, (long)G * HDIM);
  gin_gate<<<waveBlk, TPB, 0, stream>>>(bufA, gate_w, gate_b, batch, gateB, mU, N);
  gin_exp_denom<<<(N + TPB - 1) / TPB, TPB, 0, stream>>>(gateB, mU, denom, batch, N);
  gin_wsum<<<waveBlk, TPB, 0, stream>>>(bufA, gateB, denom, batch, gvec, N);

  // Heads
  const int headBlk = (G + 127) / 128;
  gin_head_wmma<<<headBlk, TPB, 0, stream>>>(gvec, Wt[4], cls_b1, cls_w2, cls_b2, out, 9, G);
  gin_head_wmma<<<headBlk, TPB, 0, stream>>>(gvec, Wt[5], reg_b1, reg_w2, reg_b2, out + (size_t)G * 9, 1, G);
}
